// SelfAttention_74887049773772
// MI455X (gfx1250) — matrix operand order repaired and checked by reading, not compiled, not run on hardware
//
#include <hip/hip_runtime.h>
#include <hip/hip_bf16.h>

typedef __attribute__((ext_vector_type(16))) _Float16 v16h;
typedef __attribute__((ext_vector_type(8)))  float    v8f;

#define EMBED 512
#define HEADS 8
#define HDIM  64

// K index inside a 16x32 f16 A-fragment for element h (0..15) of the v16h,
// contiguous order: lanes 0-15 hold K {0..15}, lanes 16-31 hold K {16..31}.
static __device__ __forceinline__ int a_k_of(int lane, int h) {
    int k = (lane >= 16) ? 16 : 0;
    k += h;
    return k;
}

// All-reduce max over each 16-lane row of the wave (both halves independently).
// DPP16 row_ror rotate-combine: after steps 1,2,4,8 every lane holds the row max.
static __device__ __forceinline__ float rowmax16(float v) {
#if __has_builtin(__builtin_amdgcn_update_dpp)
    int x, y;
    x = __float_as_int(v);
    y = __builtin_amdgcn_update_dpp(x, x, 0x121, 0xF, 0xF, false);  // row_ror:1
    v = fmaxf(v, __int_as_float(y));
    x = __float_as_int(v);
    y = __builtin_amdgcn_update_dpp(x, x, 0x122, 0xF, 0xF, false);  // row_ror:2
    v = fmaxf(v, __int_as_float(y));
    x = __float_as_int(v);
    y = __builtin_amdgcn_update_dpp(x, x, 0x124, 0xF, 0xF, false);  // row_ror:4
    v = fmaxf(v, __int_as_float(y));
    x = __float_as_int(v);
    y = __builtin_amdgcn_update_dpp(x, x, 0x128, 0xF, 0xF, false);  // row_ror:8
    v = fmaxf(v, __int_as_float(y));
    return v;
#else
    v = fmaxf(v, __shfl_xor(v, 1, 32));
    v = fmaxf(v, __shfl_xor(v, 2, 32));
    v = fmaxf(v, __shfl_xor(v, 4, 32));
    v = fmaxf(v, __shfl_xor(v, 8, 32));
    return v;
#endif
}

__global__ void cvt_f32_to_f16_kernel(const float* __restrict__ src,
                                      _Float16* __restrict__ dst, int n) {
    int i = blockIdx.x * blockDim.x + threadIdx.x;
    if (i < n) dst[i] = (_Float16)src[i];
}

// Per-head projection: Y[row, o] = oscale * sum_d X[row, d] * W[o, d]
// X: [rows, 64] f32 (rows = N*L*H, row = (nb*L + l)*H + hh), W: [64,64] f32.
// TRANSPOSE==0: store row-major [rows, 64] f16 (for Q, K)
// TRANSPOSE==1: store as [H, 64, N*L] f16 (for V; makes P@V B-loads contiguous)
template <int TRANSPOSE>
__global__ __launch_bounds__(128)
void proj_kernel(const float* __restrict__ X, const float* __restrict__ W,
                 _Float16* __restrict__ Y, int NL, float oscale) {
    int lane  = threadIdx.x & 31;
    int wave  = threadIdx.x >> 5;
    int row0  = (blockIdx.x * 4 + wave) * 16;
    int nlane = lane & 15;
    int hi8   = (lane >= 16) ? 8 : 0;
    int kb    = (lane >= 16) ? 16 : 0;

    v16h a0, a1;
    const float* xrow = X + (size_t)(row0 + nlane) * HDIM;
#pragma unroll
    for (int h = 0; h < 16; ++h) {
        int k = a_k_of(lane, h);
        a0[h] = (_Float16)xrow[k];
        a1[h] = (_Float16)xrow[32 + k];
    }
#pragma unroll
    for (int ot = 0; ot < 4; ++ot) {
        v8f c = {};
        v16h b0, b1;
        const float* wrow = W + (size_t)(ot * 16 + nlane) * HDIM;
#pragma unroll
        for (int h = 0; h < 16; ++h) {
            b0[h] = (_Float16)wrow[kb + h];
            b1[h] = (_Float16)wrow[32 + kb + h];
        }
        c = __builtin_amdgcn_wmma_f32_16x16x32_f16(false, a0, false, b0, (short)0, c, false, false);
        c = __builtin_amdgcn_wmma_f32_16x16x32_f16(false, a1, false, b1, (short)0, c, false, false);
#pragma unroll
        for (int r = 0; r < 8; ++r) {
            int orow = row0 + r + hi8;        // D layout: VGPR r -> row r (+8 for hi half)
            int ocol = ot * 16 + nlane;       // lane -> column
            _Float16 v = (_Float16)(c[r] * oscale);
            if (TRANSPOSE) {
                int hh   = orow & (HEADS - 1);  // row = lidx*H + hh
                int lidx = orow >> 3;           // lidx = nb*L + l
                Y[((size_t)hh * HDIM + ocol) * (size_t)NL + lidx] = v;
            } else {
                Y[(size_t)orow * HDIM + ocol] = v;
            }
        }
    }
}

// Flash-attention: one wave per 16-query tile for a given (n, h).
// Q is pre-scaled by 1/sqrt(512) at projection time.
// LT > 0: compile-time sequence length (row offsets become immediate offsets).
template <int LT>
__global__ __launch_bounds__(128)
void attn_kernel(const _Float16* __restrict__ qh,  // [N*L*H, 64] (pre-scaled)
                 const _Float16* __restrict__ kh,  // [N*L*H, 64]
                 const _Float16* __restrict__ vt,  // [H, 64, N*L]
                 const int* __restrict__ mask,     // [N, 1, L, L]
                 _Float16* __restrict__ ao,        // [N*L, 512]
                 int Ll_rt, int NL) {
    const int Ll = (LT > 0) ? LT : Ll_rt;
    __shared__ _Float16 p_lds[4][16][34];  // per-wave 16x32 P tile, padded

    int lane  = threadIdx.x & 31;
    int wave  = threadIdx.x >> 5;
    int hh    = blockIdx.y;
    int nb    = blockIdx.z;
    int q0    = (blockIdx.x * 4 + wave) * 16;
    int nlane = lane & 15;
    int hi8   = (lane >= 16) ? 8 : 0;
    int kb    = (lane >= 16) ? 16 : 0;
    const float MASKED = -4.4194173824159216e18f;  // -1e20 / sqrt(512)

    // Q A-fragments (16 q-rows x K=64 over d, two K-tiles)
    v16h aq0, aq1;
    {
        const _Float16* qrow = qh + ((size_t)(nb * Ll + q0 + nlane) * HEADS + hh) * HDIM;
#pragma unroll
        for (int h = 0; h < 16; ++h) {
            int k = a_k_of(lane, h);
            aq0[h] = qrow[k];
            aq1[h] = qrow[32 + k];
        }
    }

    // ones B-fragment: rowsum(P) = P @ ones via one extra WMMA
    v16h bone;
#pragma unroll
    for (int h = 0; h < 16; ++h) bone[h] = (_Float16)1.0f;

    v8f acc0 = {}, acc1 = {}, acc2 = {}, acc3 = {};
    float mrow[8], lrow[8];
#pragma unroll
    for (int r = 0; r < 8; ++r) { mrow[r] = -3.0e38f; lrow[r] = 0.0f; }

    const size_t krs = (size_t)HEADS * HDIM;  // row stride of kh in halves

    // Hoisted, strided base pointers: inner loop uses immediate offsets only.
    const _Float16* kp0 = kh + (size_t)(nb * Ll + nlane) * krs + (size_t)hh * HDIM + kb;
    const _Float16* kp1 = kp0 + 16 * krs;
    const _Float16* vbase = vt + (size_t)hh * HDIM * (size_t)NL + (size_t)nb * Ll;
    const _Float16* vp0 = vbase + (size_t)(0 * 16 + nlane) * NL + kb;
    const _Float16* vp1 = vbase + (size_t)(1 * 16 + nlane) * NL + kb;
    const _Float16* vp2 = vbase + (size_t)(2 * 16 + nlane) * NL + kb;
    const _Float16* vp3 = vbase + (size_t)(3 * 16 + nlane) * NL + kb;
    const int* mp = mask + (size_t)nb * Ll * Ll + (size_t)q0 * Ll + nlane;

    for (int kk = 0; kk < Ll; kk += 32) {
        if (kk + 32 < Ll)
            __builtin_prefetch(kp0 + (size_t)32 * krs, 0, 0);

        // scores: two 16x16 tiles (keys kk..kk+15, kk+16..kk+31)
        v8f s0 = {}, s1 = {};
        {
            v16h b = *(const v16h*)(kp0);
            s0 = __builtin_amdgcn_wmma_f32_16x16x32_f16(false, aq0, false, b, (short)0, s0, false, false);
            b = *(const v16h*)(kp0 + 32);
            s0 = __builtin_amdgcn_wmma_f32_16x16x32_f16(false, aq1, false, b, (short)0, s0, false, false);
            v16h b2 = *(const v16h*)(kp1);
            s1 = __builtin_amdgcn_wmma_f32_16x16x32_f16(false, aq0, false, b2, (short)0, s1, false, false);
            b2 = *(const v16h*)(kp1 + 32);
            s1 = __builtin_amdgcn_wmma_f32_16x16x32_f16(false, aq1, false, b2, (short)0, s1, false, false);
        }

        // mask + online max update; exp'd P staged to LDS; sums done by WMMA below
        float alpha[8];
#pragma unroll
        for (int r = 0; r < 8; ++r) {
            int m0 = mp[(size_t)(r + hi8) * Ll];
            int m1 = mp[(size_t)(r + hi8) * Ll + 16];
            float v0 = (m0 == 0) ? MASKED : s0[r];
            float v1 = (m1 == 0) ? MASKED : s1[r];
            float mt   = rowmax16(fmaxf(v0, v1));
            float mnew = fmaxf(mrow[r], mt);
            alpha[r]   = __expf(mrow[r] - mnew);
            mrow[r]    = mnew;
            float p0 = __expf(v0 - mnew);
            float p1 = __expf(v1 - mnew);
            p_lds[wave][r + hi8][nlane]      = (_Float16)p0;
            p_lds[wave][r + hi8][16 + nlane] = (_Float16)p1;
            acc0[r] *= alpha[r]; acc1[r] *= alpha[r];
            acc2[r] *= alpha[r]; acc3[r] *= alpha[r];
        }

        // reshape P (C-layout) -> A-fragment via LDS
        v16h ap;
#pragma unroll
        for (int h = 0; h < 16; ++h) {
            int k = a_k_of(lane, h);
            ap[h] = p_lds[wave][nlane][k];
        }

        // row sums: P @ ones -> D[m,n] = rowsum[m] in C-layout (matches lrow index)
        {
            v8f sums = {};
            sums = __builtin_amdgcn_wmma_f32_16x16x32_f16(false, ap, false, bone, (short)0, sums, false, false);
#pragma unroll
            for (int r = 0; r < 8; ++r) lrow[r] = lrow[r] * alpha[r] + sums[r];
        }

        // O += P @ V ; V B-fragments are contiguous along keys ([H,64,N*L] layout)
        {
            v16h b;
            b = *(const v16h*)(vp0);
            acc0 = __builtin_amdgcn_wmma_f32_16x16x32_f16(false, ap, false, b, (short)0, acc0, false, false);
            b = *(const v16h*)(vp1);
            acc1 = __builtin_amdgcn_wmma_f32_16x16x32_f16(false, ap, false, b, (short)0, acc1, false, false);
            b = *(const v16h*)(vp2);
            acc2 = __builtin_amdgcn_wmma_f32_16x16x32_f16(false, ap, false, b, (short)0, acc2, false, false);
            b = *(const v16h*)(vp3);
            acc3 = __builtin_amdgcn_wmma_f32_16x16x32_f16(false, ap, false, b, (short)0, acc3, false, false);
        }

        // advance strided pointers (constant strides; single adds per pointer)
        kp0 += 32 * krs;
        kp1 += 32 * krs;
        vp0 += 32; vp1 += 32; vp2 += 32; vp3 += 32;
        mp  += 32;
    }

    // normalize and store O tile as f16 rows of the concatenated [N*L, 512] matrix
#pragma unroll
    for (int r = 0; r < 8; ++r) {
        int q = q0 + r + hi8;
        float inv = 1.0f / lrow[r];
        size_t base = (size_t)(nb * Ll + q) * EMBED + hh * HDIM;
        ao[base + 0 * 16 + nlane] = (_Float16)(acc0[r] * inv);
        ao[base + 1 * 16 + nlane] = (_Float16)(acc1[r] * inv);
        ao[base + 2 * 16 + nlane] = (_Float16)(acc2[r] * inv);
        ao[base + 3 * 16 + nlane] = (_Float16)(acc3[r] * inv);
    }
}

// out = A @ Wo^T + bo,  A:[rows,512] f16, Wo(f16):[512,512] row-major, out f32
__global__ __launch_bounds__(128)
void outproj_kernel(const _Float16* __restrict__ A, const _Float16* __restrict__ Woh,
                    const float* __restrict__ bo, float* __restrict__ out) {
    int lane  = threadIdx.x & 31;
    int wave  = threadIdx.x >> 5;
    int r0    = blockIdx.x * 16;
    int c0    = (blockIdx.y * 4 + wave) * 16;
    int nlane = lane & 15;
    int hi8   = (lane >= 16) ? 8 : 0;
    int kb    = (lane >= 16) ? 16 : 0;

    v8f c = {};
    const _Float16* arow = A + (size_t)(r0 + nlane) * EMBED;
    const _Float16* wrow = Woh + (size_t)(c0 + nlane) * EMBED;
#pragma unroll
    for (int kt = 0; kt < EMBED; kt += 32) {
        v16h a;
#pragma unroll
        for (int h = 0; h < 16; ++h) a[h] = arow[kt + a_k_of(lane, h)];
        v16h b = *(const v16h*)(wrow + kt + kb);
        c = __builtin_amdgcn_wmma_f32_16x16x32_f16(false, a, false, b, (short)0, c, false, false);
    }
#pragma unroll
    for (int r = 0; r < 8; ++r) {
        int orow = r0 + r + hi8;
        int ocol = c0 + nlane;
        out[(size_t)orow * EMBED + ocol] = c[r] + bo[ocol];
    }
}

extern "C" void kernel_launch(void* const* d_in, const int* in_sizes, int n_in,
                              void* d_out, int out_size, void* d_ws, size_t ws_size,
                              hipStream_t stream) {
    const float* values = (const float*)d_in[0];
    const float* keys   = (const float*)d_in[1];
    const float* query  = (const float*)d_in[2];
    const int*   mask   = (const int*)d_in[3];
    const float* Wv     = (const float*)d_in[4];
    const float* Wk     = (const float*)d_in[5];
    const float* Wq     = (const float*)d_in[6];
    const float* Wo     = (const float*)d_in[7];
    const float* bo     = (const float*)d_in[8];
    float*       out    = (float*)d_out;

    long long tot = in_sizes[0];              // N*L*512
    long long msz = in_sizes[3];              // N*L*L
    int L  = (int)((msz * 512) / tot);
    int N  = (int)(tot / ((long long)L * EMBED));
    int NL = N * L;

    size_t S = (size_t)NL * EMBED * sizeof(_Float16);  // one f16 activation buffer
    char* ws = (char*)d_ws;
    _Float16* qh  = (_Float16*)(ws + 0 * S);
    _Float16* kh  = (_Float16*)(ws + 1 * S);
    _Float16* vt  = (_Float16*)(ws + 2 * S);
    _Float16* ao  = (_Float16*)(ws + 3 * S);
    _Float16* woh = (_Float16*)(ws + 4 * S);

    const float qscale = 0.044194173824159216f;  // 1/sqrt(512)

    // 1) Wo -> f16
    {
        int n = EMBED * EMBED;
        cvt_f32_to_f16_kernel<<<(n + 255) / 256, 256, 0, stream>>>(Wo, woh, n);
    }
    // 2) per-head projections (rows = N*L*H of 64-dim vectors)
    {
        int rows = NL * HEADS;
        int blocks = rows / 64;  // 4 waves/block * 16 rows/wave
        proj_kernel<1><<<blocks, 128, 0, stream>>>(values, Wv, vt, NL, 1.0f);
        proj_kernel<0><<<blocks, 128, 0, stream>>>(keys,   Wk, kh, NL, 1.0f);
        proj_kernel<0><<<blocks, 128, 0, stream>>>(query,  Wq, qh, NL, qscale);
    }
    // 3) flash attention (compile-time L fast path)
    dim3 agrid(L / 64, HEADS, N);
    if (L == 2048) {
        attn_kernel<2048><<<agrid, 128, 0, stream>>>(qh, kh, vt, mask, ao, L, NL);
    } else {
        attn_kernel<0><<<agrid, 128, 0, stream>>>(qh, kh, vt, mask, ao, L, NL);
    }
    // 4) output projection + bias
    outproj_kernel<<<dim3(NL / 16, EMBED / 64), 128, 0, stream>>>(ao, woh, bo, out);
}
